// VectorQuantizerP_84980222919422
// MI455X (gfx1250) — matrix operand order repaired and checked by reading, not compiled, not run on hardware
//
#include <hip/hip_runtime.h>
#include <hip/hip_bf16.h>
#include <stdint.h>

typedef __bf16 bf16;
typedef bf16  v16bf __attribute__((ext_vector_type(16)));
typedef bf16  v8bf  __attribute__((ext_vector_type(8)));
typedef float v8f   __attribute__((ext_vector_type(8)));
typedef float v4f   __attribute__((ext_vector_type(4)));
typedef int   v4i   __attribute__((ext_vector_type(4)));

#define N_TOK 32768
#define K_CB  8192
#define D_DIM 256
#define KSPLIT 4
#define KT_PER_CHUNK (K_CB / 16 / KSPLIT)   // 128 k-tiles per block

// ---- CDNA5 async global->LDS path (guarded so compile never breaks) --------
#if __has_builtin(__builtin_amdgcn_global_load_async_to_lds_b128) && \
    __has_builtin(__builtin_amdgcn_s_wait_asynccnt)
#define USE_ASYNC_LDS 1
#else
#define USE_ASYNC_LDS 0
#endif

// ---------------------------------------------------------------------------
// K1: codebook = embedding @ proj_w^T + proj_b ; also bf16 copy and ||c||^2
// ---------------------------------------------------------------------------
__global__ __launch_bounds__(256)
void codebook_kernel(const float* __restrict__ emb, const float* __restrict__ W,
                     const float* __restrict__ bias, float* __restrict__ cb_f32,
                     bf16* __restrict__ cb_bf, float* __restrict__ cnorm) {
    __shared__ float e[D_DIM];
    __shared__ float red[256];
    const int k = blockIdx.x;
    const int d = threadIdx.x;
    e[d] = emb[(size_t)k * D_DIM + d];
    __syncthreads();
    const float* wrow = W + (size_t)d * D_DIM;
    float s = bias[d];
    #pragma unroll 8
    for (int j = 0; j < D_DIM; ++j) s = fmaf(e[j], wrow[j], s);
    cb_f32[(size_t)k * D_DIM + d] = s;
    cb_bf [(size_t)k * D_DIM + d] = (bf16)s;
    red[d] = s * s;
    __syncthreads();
    for (int off = 128; off > 0; off >>= 1) {
        if (d < off) red[d] += red[d + off];
        __syncthreads();
    }
    if (d == 0) cnorm[k] = red[0];
}

// ---------------------------------------------------------------------------
// K2: fused distance GEMM + partial argmin.
// score(n,k) = ||c_k||^2 - 2 * z_n . c_k   (||z_n||^2 constant per row -> drop)
// Block = 8 waves; wave owns TWO 16-row A tiles (register resident) so each
// B fragment from LDS feeds two v_wmma_f32_16x16x32_bf16 (halves LDS traffic,
// two independent accumulation chains). Codebook tiles double-buffered in LDS
// via async global->LDS loads (ASYNCcnt). K split 4-ways across blocks for
// grid parallelism; partial argmins combined in K2b.
// ---------------------------------------------------------------------------
__global__ __launch_bounds__(256)
void argmin_kernel(const float* __restrict__ z, const bf16* __restrict__ cb_bf,
                   const float* __restrict__ cnorm,
                   unsigned long long* __restrict__ bestOut) {
    __shared__ __align__(16) bf16 tileB[2][16 * D_DIM];   // 2 x 8 KB
    const int tid      = threadIdx.x;
    const int wave     = tid >> 5;
    const int lane     = tid & 31;
    const int lane16   = lane & 15;
    const int laneHalf = lane >> 4;
    const int rowBlk   = blockIdx.x / KSPLIT;             // 0..127
    const int kChunk   = blockIdx.x % KSPLIT;             // 0..3
    const int rowBase0 = rowBlk * 256 + wave * 16;
    const int rowBase1 = rowBase0 + 128;
    const int kStart   = kChunk * (K_CB / KSPLIT);

    // ---- A fragments (z -> bf16), ISA 16-bit A 16x32 layout:
    // lanes 0-15 hold K = {0..7, 16..23}; lanes 16-31 hold K = {8..15, 24..31}
    v16bf afrag0[8], afrag1[8];
    #pragma unroll
    for (int t = 0; t < 2; ++t) {
        const int rb = t ? rowBase1 : rowBase0;
        const float* zr = z + (size_t)(rb + lane16) * D_DIM;
        #pragma unroll
        for (int c = 0; c < 8; ++c) {
            const int d0 = 32 * c + 8 * laneHalf;
            const int d1 = 32 * c + 16 + 8 * laneHalf;
            v4f f0 = *(const v4f*)(zr + d0);
            v4f f1 = *(const v4f*)(zr + d0 + 4);
            v4f f2 = *(const v4f*)(zr + d1);
            v4f f3 = *(const v4f*)(zr + d1 + 4);
            v16bf a;
            #pragma unroll
            for (int i = 0; i < 4; ++i) {
                a[i] = (bf16)f0[i]; a[4 + i] = (bf16)f1[i];
                a[8 + i] = (bf16)f2[i]; a[12 + i] = (bf16)f3[i];
            }
            if (t) afrag1[c] = a; else afrag0[c] = a;
        }
    }

    unsigned long long best0[8], best1[8];
    #pragma unroll
    for (int r = 0; r < 8; ++r) { best0[r] = ~0ull; best1[r] = ~0ull; }

    // stage k-tile `kt` (16 codebook rows, 8192 B = 512 x 16B) into buffer `b`
    auto stage = [&](int kt, int b) {
#if USE_ASYNC_LDS
        __attribute__((address_space(1))) v4i* gsrc =
            (__attribute__((address_space(1))) v4i*)
                (cb_bf + (size_t)(kStart + kt * 16) * D_DIM);
        __attribute__((address_space(3))) v4i* ldst =
            (__attribute__((address_space(3))) v4i*)(&tileB[b][0]);
        __builtin_amdgcn_global_load_async_to_lds_b128(
            gsrc + tid,       ldst + tid,       0, 0);
        __builtin_amdgcn_global_load_async_to_lds_b128(
            gsrc + tid + 256, ldst + tid + 256, 0, 0);
#else
        const uint4* s4 =
            (const uint4*)(cb_bf + (size_t)(kStart + kt * 16) * D_DIM);
        uint4* d4 = (uint4*)&tileB[b][0];
        d4[tid]       = s4[tid];
        d4[tid + 256] = s4[tid + 256];
#endif
    };

    stage(0, 0);                           // prologue: tile 0 in flight
    for (int kt = 0; kt < KT_PER_CHUNK; ++kt) {
        const int cur = kt & 1;
        if (kt + 1 < KT_PER_CHUNK) {       // prefetch next tile (uniform branch)
            stage(kt + 1, cur ^ 1);
#if USE_ASYNC_LDS
            // in-order completion: leave only the just-issued pair outstanding
            __builtin_amdgcn_s_wait_asynccnt(2);
        } else {
            __builtin_amdgcn_s_wait_asynccnt(0);
#endif
        }
        __syncthreads();                   // tile kt resident for all waves

        const int kBase = kStart + kt * 16;
        const float cn = cnorm[kBase + lane16];
        v8f acc0 = {0.f,0.f,0.f,0.f,0.f,0.f,0.f,0.f};
        v8f acc1 = {0.f,0.f,0.f,0.f,0.f,0.f,0.f,0.f};
        #pragma unroll
        for (int c = 0; c < 8; ++c) {
            // ISA 16-bit B 32x16 layout: lanes 0-15 K={0..7,16..23}, lanes 16-31
            // K={8..15,24..31}; column N = lane16 = codebook row within tile.
            const bf16* bp = &tileB[cur][lane16 * D_DIM + 32 * c + 8 * laneHalf];
            v8bf blo = *(const v8bf*)(bp);
            v8bf bhi = *(const v8bf*)(bp + 16);
            v16bf bfrag;
            #pragma unroll
            for (int i = 0; i < 8; ++i) { bfrag[i] = blo[i]; bfrag[8 + i] = bhi[i]; }
            acc0 = __builtin_amdgcn_wmma_f32_16x16x32_bf16(
                false, afrag0[c], false, bfrag, (short)0, acc0, false, false);
            acc1 = __builtin_amdgcn_wmma_f32_16x16x32_bf16(
                false, afrag1[c], false, bfrag, (short)0, acc1, false, false);
        }

        // C layout: VGPR r -> (M = r + 8*laneHalf, N = lane16)
        #pragma unroll
        for (int r = 0; r < 8; ++r) {
            float s0 = fmaf(-2.0f, acc0[r], cn);
            float s1 = fmaf(-2.0f, acc1[r], cn);
            unsigned u0 = __float_as_uint(s0);
            unsigned u1 = __float_as_uint(s1);
            u0 = (u0 & 0x80000000u) ? ~u0 : (u0 | 0x80000000u);
            u1 = (u1 & 0x80000000u) ? ~u1 : (u1 | 0x80000000u);
            unsigned long long k0 =
                ((unsigned long long)u0 << 32) | (unsigned)(kBase + lane16);
            unsigned long long k1 =
                ((unsigned long long)u1 << 32) | (unsigned)(kBase + lane16);
            best0[r] = (k0 < best0[r]) ? k0 : best0[r];
            best1[r] = (k1 < best1[r]) ? k1 : best1[r];
        }
        __syncthreads();                   // done reading buf before reuse
    }

    // min-reduce across the 16 lanes that share a row; lane16==0 writes.
    #pragma unroll
    for (int t = 0; t < 2; ++t) {
        const int rb = t ? rowBase1 : rowBase0;
        #pragma unroll
        for (int r = 0; r < 8; ++r) {
            unsigned long long b = t ? best1[r] : best0[r];
            #pragma unroll
            for (int m = 1; m < 16; m <<= 1) {
                unsigned long long o = __shfl_xor(b, m, 32);
                b = (o < b) ? o : b;
            }
            if (lane16 == 0)
                bestOut[(size_t)(rb + r + 8 * laneHalf) * KSPLIT + kChunk] = b;
        }
    }
}

// K2b: combine the KSPLIT partial argmins per row (lowest-index tie-break is
// preserved by the packed key ordering).
__global__ __launch_bounds__(256)
void combine_kernel(const unsigned long long* __restrict__ bestOut,
                    int* __restrict__ indices) {
    const int n = blockIdx.x * 256 + threadIdx.x;
    if (n < N_TOK) {
        unsigned long long b = bestOut[(size_t)n * KSPLIT];
        #pragma unroll
        for (int j = 1; j < KSPLIT; ++j) {
            unsigned long long o = bestOut[(size_t)n * KSPLIT + j];
            b = (o < b) ? o : b;
        }
        indices[n] = (int)(b & 0xFFFFFFFFull);
    }
}

// ---------------------------------------------------------------------------
// K3: gather zq = codebook[idx] -> out, indices as float, loss partials.
// ---------------------------------------------------------------------------
__global__ __launch_bounds__(256)
void gather_kernel(const float* __restrict__ z, const float* __restrict__ cb_f32,
                   const int* __restrict__ indices, float* __restrict__ out,
                   float* __restrict__ partials) {
    __shared__ float red[256];
    const int tid = threadIdx.x;
    float acc = 0.f;
    const size_t total = (size_t)N_TOK * D_DIM;
    for (size_t e = (size_t)blockIdx.x * 256 + tid; e < total;
         e += (size_t)gridDim.x * 256) {
        const int n = (int)(e >> 8);
        const int d = (int)(e & 255);
        const float q = cb_f32[(size_t)indices[n] * D_DIM + d];
        out[e] = q;
        const float diff = q - z[e];
        acc = fmaf(diff, diff, acc);
    }
    for (int n = blockIdx.x * 256 + tid; n < N_TOK; n += gridDim.x * 256)
        out[total + n] = (float)indices[n];
    red[tid] = acc;
    __syncthreads();
    for (int off = 128; off > 0; off >>= 1) {
        if (tid < off) red[tid] += red[tid + off];
        __syncthreads();
    }
    if (tid == 0) partials[blockIdx.x] = red[0];
}

// loss = beta*mean((zq-z)^2) + mean((zq-z)^2) = 1.5 * sum / (N*D)
__global__ __launch_bounds__(256)
void loss_kernel(const float* __restrict__ partials, int nPart,
                 float* __restrict__ out) {
    __shared__ float red[256];
    float a = 0.f;
    for (int i = threadIdx.x; i < nPart; i += 256) a += partials[i];
    red[threadIdx.x] = a;
    __syncthreads();
    for (int off = 128; off > 0; off >>= 1) {
        if (threadIdx.x < off) red[threadIdx.x] += red[threadIdx.x + off];
        __syncthreads();
    }
    if (threadIdx.x == 0)
        out[(size_t)N_TOK * D_DIM + N_TOK] =
            1.5f * red[0] / (float)((size_t)N_TOK * D_DIM);
}

// ---------------------------------------------------------------------------
extern "C" void kernel_launch(void* const* d_in, const int* in_sizes, int n_in,
                              void* d_out, int out_size, void* d_ws, size_t ws_size,
                              hipStream_t stream) {
    (void)in_sizes; (void)n_in; (void)out_size; (void)ws_size;
    const float* z    = (const float*)d_in[0];   // [N, D]
    const float* emb  = (const float*)d_in[1];   // [K, D]
    const float* W    = (const float*)d_in[2];   // [D, D]
    const float* bias = (const float*)d_in[3];   // [D]
    float* out = (float*)d_out;                  // zq | indices(as f32) | loss

    char* ws = (char*)d_ws;
    float* cb_f32   = (float*)(ws);                                    // 8 MB
    bf16*  cb_bf    = (bf16*) (ws + 8ull * 1024 * 1024);               // 4 MB
    float* cnorm    = (float*)(ws + 12ull * 1024 * 1024);              // 32 KB
    int*   idx      = (int*)  (ws + 12ull * 1024 * 1024 + 64 * 1024);  // 128 KB
    float* partials = (float*)(ws + 12ull * 1024 * 1024 + 256 * 1024); // 2 KB
    unsigned long long* bestOut =
        (unsigned long long*)(ws + 13ull * 1024 * 1024);               // 1 MB

    codebook_kernel<<<K_CB, 256, 0, stream>>>(emb, W, bias, cb_f32, cb_bf, cnorm);
    argmin_kernel<<<(N_TOK / 256) * KSPLIT, 256, 0, stream>>>(z, cb_bf, cnorm, bestOut);
    combine_kernel<<<N_TOK / 256, 256, 0, stream>>>(bestOut, idx);
    gather_kernel<<<512, 256, 0, stream>>>(z, cb_f32, idx, out, partials);
    loss_kernel<<<1, 256, 0, stream>>>(partials, 512, out);
}
